// TemplatePairStackIteration_62663572848698
// MI455X (gfx1250) — matrix operand order repaired and checked by reading, not compiled, not run on hardware
//
#include <hip/hip_runtime.h>
#include <hip/hip_fp16.h>

// ---------------------------------------------------------------------------
// TemplatePairStackIteration for gfx1250 (MI455X, wave32, WMMA 16x16x32 f16)
//   x = x + tri_attn_start(x); x = x + tri_attn_end(x);
//   x = x + tri_mult_out(x);   x = x + tri_mult_in(x);
// Contractions on v_wmma_f32_16x16x32_f16; LDS staging via async-to-LDS DMA
// (ASYNCcnt) where layouts allow; LN/softmax/sigmoid on VALU.
// ---------------------------------------------------------------------------

typedef __attribute__((ext_vector_type(16))) _Float16 v16h;
typedef __attribute__((ext_vector_type(8)))  float    v8f;

struct __align__(16) f4raw { float x, y, z, w; };
struct __align__(8)  f2raw { float x, y; };

union AFrag { v16h v; f4raw f4[2]; };

#define NNPX 256
#define CCH  128
#define PPOS 65536

__device__ static inline v8f wmma16(v16h a, v16h b, v8f c) {
  // D = A(16x32 f16) * B(32x16 f16) + C(16x16 f32)
  return __builtin_amdgcn_wmma_f32_16x16x32_f16(false, a, false, b, (short)0, c,
                                                false, false);
}

// Async DMA: 16 bytes global -> LDS, tracked by ASYNCcnt (no VGPR data path).
// Generic LDS pointer low 32 bits == LDS byte offset (aperture rule).
__device__ static inline void async_load_b128(unsigned lds_off, const void* g) {
  asm volatile("global_load_async_to_lds_b128 %0, %1, off"
               :: "v"(lds_off), "v"(g) : "memory");
}
__device__ static inline void wait_async0() {
  asm volatile("s_wait_asynccnt 0x0" ::: "memory");
}

// A-matrix fragment: 16 rows x 32 K, row-major source, rowStride in halfs.
// Layout: lane(hi=0): halves 0..15 = K0..15;
//         lane(hi=1): halves 0..15 = K16..31.
__device__ static inline v16h load_a_frag(const _Float16* base, int rowStride, int lane) {
  int m = lane & 15, hi = lane >> 4;
  const _Float16* row = base + (size_t)m * rowStride + hi * 16;
  AFrag u;
  u.f4[0] = *(const f4raw*)(row);
  u.f4[1] = *(const f4raw*)(row + 8);
  return u.v;
}

// B-matrix fragment: 32 K x 16 cols, from column-major storage Bt[col][K]
// (i.e. transposed weights), rowStride in halfs.
// Layout (same k order as load_a_frag): lane n (hi=0) holds col n, K=0..15; lane n+16 holds col n, K=16..31.
__device__ static inline v16h load_b_frag(const _Float16* baseT, int rowStride, int lane) {
  int n = lane & 15, hi = lane >> 4;
  const _Float16* col = baseT + (size_t)n * rowStride + hi * 16;
  AFrag u;
  u.f4[0] = *(const f4raw*)(col);
  u.f4[1] = *(const f4raw*)(col + 8);
  return u.v;
}

__device__ static inline float sigm(float x) { return 1.0f / (1.0f + __expf(-x)); }

// ---------------------------------------------------------------------------
// LayerNorm over last dim (128) of [rows,128] fp32 -> f16.  One wave per row.
// ---------------------------------------------------------------------------
__global__ __launch_bounds__(32) void ln_kernel(const float* __restrict__ x,
                                                const float* __restrict__ s,
                                                const float* __restrict__ b,
                                                _Float16* __restrict__ out) {
  int row = blockIdx.x, lane = threadIdx.x;
  f4raw v = ((const f4raw*)(x + (size_t)row * CCH))[lane];
  float sum = v.x + v.y + v.z + v.w;
  for (int o = 16; o; o >>= 1) sum += __shfl_xor(sum, o);
  float mu = sum * (1.0f / 128.0f);
  float dx = v.x - mu, dy = v.y - mu, dz = v.z - mu, dw = v.w - mu;
  float ss = dx * dx + dy * dy + dz * dz + dw * dw;
  for (int o = 16; o; o >>= 1) ss += __shfl_xor(ss, o);
  float rs = rsqrtf(ss * (1.0f / 128.0f) + 1e-5f);
  union { f2raw f2; _Float16 hh[4]; } pack;
  int cb = lane * 4;
  pack.hh[0] = (_Float16)(dx * rs * s[cb + 0] + b[cb + 0]);
  pack.hh[1] = (_Float16)(dy * rs * s[cb + 1] + b[cb + 1]);
  pack.hh[2] = (_Float16)(dz * rs * s[cb + 2] + b[cb + 2]);
  pack.hh[3] = (_Float16)(dw * rs * s[cb + 3] + b[cb + 3]);
  ((f2raw*)(out + (size_t)row * CCH))[lane] = pack.f2;
}

// ---------------------------------------------------------------------------
// Weight transpose + f16 convert: W[K][N] f32 -> Wt[N][K] f16.
// ---------------------------------------------------------------------------
__global__ void wt_kernel(const float* __restrict__ W, _Float16* __restrict__ Wt,
                          int K, int Nn) {
  int idx = blockIdx.x * 256 + threadIdx.x;
  if (idx >= K * Nn) return;
  int n = idx / K, k = idx - n * K;
  Wt[idx] = (_Float16)W[(size_t)k * Nn + n];
}

// ---------------------------------------------------------------------------
// Attention projections: qkv (384 cols, scattered to [h][i][j][a] f16) and
// gate g = sigmoid(xn@Wg+bg) (128 cols).  One wave per 16x16 output tile.
// grid = (4096, 32): y<24 -> qkv tile, y>=24 -> g tile.
// ---------------------------------------------------------------------------
__global__ __launch_bounds__(32) void attn_proj_kernel(
    const _Float16* __restrict__ xn, const _Float16* __restrict__ Wqkv_t,
    const _Float16* __restrict__ Wg_t, const float* __restrict__ bg,
    _Float16* __restrict__ q_t, _Float16* __restrict__ k_t,
    _Float16* __restrict__ v_t, _Float16* __restrict__ g16) {
  const int lane = threadIdx.x;
  const int p0 = blockIdx.x * 16;
  const int ct = blockIdx.y;
  const int n = lane & 15, hi = lane >> 4;
  const bool isg = ct >= 24;
  const int colbase = isg ? (ct - 24) * 16 : ct * 16;
  const _Float16* Wt = isg ? Wg_t : Wqkv_t;
  v8f acc{};
  for (int kk = 0; kk < 4; ++kk) {
    v16h a = load_a_frag(xn + (size_t)p0 * CCH + kk * 32, CCH, lane);
    v16h b = load_b_frag(Wt + (size_t)colbase * CCH + kk * 32, CCH, lane);
    acc = wmma16(a, b, acc);
  }
  int col = colbase + n;
  if (isg) {
    float bgv = bg[col];
    for (int r = 0; r < 8; ++r) {
      int p = p0 + hi * 8 + r;
      g16[(size_t)p * CCH + col] = (_Float16)sigm(acc[r] + bgv);
    }
  } else {
    // Wqkv flat col = a*12 + part*4 + h   (part: 0=q 1=k 2=v)
    int a_ = col / 12, rem = col % 12, part = rem >> 2, hh = rem & 3;
    _Float16* dst = (part == 0) ? q_t : ((part == 1) ? k_t : v_t);
    for (int r = 0; r < 8; ++r) {
      int p = p0 + hi * 8 + r;
      int i = p >> 8, j = p & 255;
      dst[((((size_t)hh * NNPX + i) * NNPX + j) * 32) + a_] = (_Float16)acc[r];
    }
  }
}

// bias[b,i,j,h] = xn . Wb[:,h]  -> bias_t[h][i*256+j]   (tiny GEMV, VALU)
__global__ void attn_bias_kernel(const _Float16* __restrict__ xn,
                                 const float* __restrict__ Wb,
                                 float* __restrict__ bias_t) {
  int idx = blockIdx.x * 256 + threadIdx.x;  // PPOS*4 threads
  int p = idx >> 2, h = idx & 3;
  float acc = 0.f;
  const f4raw* xr = (const f4raw*)(xn + (size_t)p * CCH);
  for (int cc = 0; cc < 16; ++cc) {  // 8 halfs per 16B chunk
    union { f4raw f4; _Float16 hh[8]; } u;
    u.f4 = xr[cc];
#pragma unroll
    for (int q = 0; q < 8; ++q) acc += (float)u.hh[q] * Wb[(cc * 8 + q) * 4 + h];
  }
  bias_t[(size_t)h * PPOS + p] = acc;
}

// ---------------------------------------------------------------------------
// Triangle attention core: one block per (line, head).
//   starting=1: fixed i=line, attend over k: S[j,k]=q_ij.k_ik + bias[k,j,h]
//   starting=0: fixed j=line, attend over k: S[i,k]=q_ij.k_kj + bias[i,k,h]
// Q/K staged to LDS with async-to-LDS DMA; V transposed manually.
// Flash-style online softmax computed in transposed score domain
// (S^T rows = keys), O^T accumulated via WMMA against V^T.
// ---------------------------------------------------------------------------
__global__ __launch_bounds__(256) void attn_core_kernel(
    const _Float16* __restrict__ q_t, const _Float16* __restrict__ k_t,
    const _Float16* __restrict__ v_t, const float* __restrict__ bias_t,
    const _Float16* __restrict__ g16, _Float16* __restrict__ o16, int starting) {
  __shared__ __align__(16) _Float16 sQ[NNPX][32];
  __shared__ __align__(16) _Float16 sK[NNPX][32];
  __shared__ __align__(16) _Float16 sVt[32][NNPX];
  const int line = blockIdx.x, h = blockIdx.y;
  const int tid = threadIdx.x;
  {  // stage: thread t handles row t; Q/K via async DMA, V via transpose
    int rrow = tid;
    size_t base = starting ? ((((size_t)h * NNPX + line) * NNPX + rrow) * 32)
                           : ((((size_t)h * NNPX + rrow) * NNPX + line) * 32);
    unsigned qoff = (unsigned)(unsigned long long)(void*)&sQ[rrow][0];
    unsigned koff = (unsigned)(unsigned long long)(void*)&sK[rrow][0];
#pragma unroll
    for (int s2 = 0; s2 < 4; ++s2) {
      async_load_b128(qoff + s2 * 16, q_t + base + s2 * 8);
      async_load_b128(koff + s2 * 16, k_t + base + s2 * 8);
    }
    const f4raw* vs = (const f4raw*)(v_t + base);
    union { f4raw f4[4]; _Float16 hh[32]; } vv;
    for (int s2 = 0; s2 < 4; ++s2) vv.f4[s2] = vs[s2];
    for (int a2 = 0; a2 < 32; ++a2) sVt[a2][rrow] = vv.hh[a2];
    wait_async0();
  }
  __syncthreads();
  const int wave = tid >> 5, lane = tid & 31;
  const int n = lane & 15, hi2 = lane >> 4;
  const float scale = 0.17677669529663687f;  // 1/sqrt(A=32)
  const float* bh = bias_t + (size_t)h * PPOS;
  for (int jj = 0; jj < 2; ++jj) {
    const int jt = wave + jj * 8;            // 16-query tile
    const int qidx = jt * 16 + n;
    __builtin_prefetch(bh + (starting ? (size_t)qidx : (size_t)qidx * NNPX), 0, 1);
    v16h bq = load_b_frag(&sQ[jt * 16][0], 32, lane);  // B = Q^T (a x j)
    float m = -3.0e38f, l = 0.0f;
    v8f oa{}, ob{};                           // O^T accum: a 0..15 / 16..31
    for (int kc = 0; kc < 8; ++kc) {          // key chunks of 32
      v16h a0 = load_a_frag(&sK[kc * 32][0], 32, lane);
      v16h a1 = load_a_frag(&sK[kc * 32 + 16][0], 32, lane);
      v8f s0{}, s1{};
      s0 = wmma16(a0, bq, s0);                // S^T tile: rows k, cols j
      s1 = wmma16(a1, bq, s1);
      float p0[8], p1[8];
      float tmax = -3.0e38f;
#pragma unroll
      for (int r = 0; r < 8; ++r) {
        int k0 = kc * 32 + hi2 * 8 + r;
        int k1 = k0 + 16;
        float b0 = starting ? bh[k0 * NNPX + qidx] : bh[qidx * NNPX + k0];
        float b1 = starting ? bh[k1 * NNPX + qidx] : bh[qidx * NNPX + k1];
        p0[r] = s0[r] * scale + b0;
        p1[r] = s1[r] * scale + b1;
        tmax = fmaxf(tmax, fmaxf(p0[r], p1[r]));
      }
      tmax = fmaxf(tmax, __shfl_xor(tmax, 16));  // combine row halves
      float mnew = fmaxf(m, tmax);
      float corr = __expf(m - mnew);
      float rsum = 0.f;
#pragma unroll
      for (int r = 0; r < 8; ++r) {
        p0[r] = __expf(p0[r] - mnew);
        p1[r] = __expf(p1[r] - mnew);
        rsum += p0[r] + p1[r];
      }
      rsum += __shfl_xor(rsum, 16);
      l = l * corr + rsum;
      m = mnew;
#pragma unroll
      for (int r = 0; r < 8; ++r) { oa[r] *= corr; ob[r] *= corr; }
      // assemble probability B-frag (32k x 16j) from the two D tiles
      v16h pf{};
#pragma unroll
      for (int r = 0; r < 8; ++r) {
        float q0 = __shfl_xor(p0[r], 16);
        float q1 = __shfl_xor(p1[r], 16);
        pf[r]     = (_Float16)(hi2 ? q1 : p0[r]);
        pf[r + 8] = (_Float16)(hi2 ? p1[r] : q0);
      }
      v16h av0 = load_a_frag(&sVt[0][kc * 32], NNPX, lane);   // V^T a 0..15
      v16h av1 = load_a_frag(&sVt[16][kc * 32], NNPX, lane);  // V^T a 16..31
      oa = wmma16(av0, pf, oa);
      ob = wmma16(av1, pf, ob);
    }
    float linv = 1.0f / l;
    size_t p = starting ? ((size_t)line * NNPX + qidx) : ((size_t)qidx * NNPX + line);
#pragma unroll
    for (int r = 0; r < 8; ++r) {
      int a0 = hi2 * 8 + r, a1 = a0 + 16;
      float g0 = (float)g16[p * CCH + a0 * 4 + h];
      float g1 = (float)g16[p * CCH + a1 * 4 + h];
      o16[p * CCH + a0 * 4 + h] = (_Float16)(oa[r] * linv * g0);
      o16[p * CCH + a1 * 4 + h] = (_Float16)(ob[r] * linv * g1);
    }
  }
}

// out-projection + residual: xdst = xsrc + o16 @ Wo + bo
__global__ __launch_bounds__(32) void attn_out_kernel(
    const _Float16* __restrict__ o16, const _Float16* __restrict__ Wo_t,
    const float* __restrict__ bo, const float* __restrict__ xsrc,
    float* __restrict__ xdst) {
  int lane = threadIdx.x, p0 = blockIdx.x * 16, ct = blockIdx.y;
  int n = lane & 15, hi = lane >> 4;
  v8f acc{};
  for (int kk = 0; kk < 4; ++kk) {
    v16h a = load_a_frag(o16 + (size_t)p0 * CCH + kk * 32, CCH, lane);
    v16h b = load_b_frag(Wo_t + (size_t)(ct * 16) * CCH + kk * 32, CCH, lane);
    acc = wmma16(a, b, acc);
  }
  int col = ct * 16 + n;
  float bv = bo[col];
  for (int r = 0; r < 8; ++r) {
    size_t p = p0 + hi * 8 + r;
    xdst[p * CCH + col] = xsrc[p * CCH + col] + acc[r] + bv;
  }
}

// ---------------------------------------------------------------------------
// Triangle-mult projection: out = (xn@W + b) * sigmoid(xn@Wg + bg),
// stored CHANNEL-MAJOR f16: out_t[c][p]  (p = i*256+k) for the per-channel
// 256x256 matmuls.
// ---------------------------------------------------------------------------
__global__ __launch_bounds__(32) void mult_proj_kernel(
    const _Float16* __restrict__ xn, const _Float16* __restrict__ W_t,
    const float* __restrict__ bb, const _Float16* __restrict__ Wg_t,
    const float* __restrict__ bgg, _Float16* __restrict__ out_t) {
  int lane = threadIdx.x, p0 = blockIdx.x * 16, ct = blockIdx.y;
  int n = lane & 15, hi = lane >> 4;
  v8f acc{}, accg{};
  for (int kk = 0; kk < 4; ++kk) {
    v16h a = load_a_frag(xn + (size_t)p0 * CCH + kk * 32, CCH, lane);
    v16h b1 = load_b_frag(W_t + (size_t)(ct * 16) * CCH + kk * 32, CCH, lane);
    v16h b2 = load_b_frag(Wg_t + (size_t)(ct * 16) * CCH + kk * 32, CCH, lane);
    acc = wmma16(a, b1, acc);
    accg = wmma16(a, b2, accg);
  }
  int col = ct * 16 + n;
  float bv = bb[col], bgv = bgg[col];
  union { f4raw f4; _Float16 hh[8]; } pack;
#pragma unroll
  for (int r = 0; r < 8; ++r)
    pack.hh[r] = (_Float16)((acc[r] + bv) * sigm(accg[r] + bgv));
  *(f4raw*)(out_t + (size_t)col * PPOS + p0 + hi * 8) = pack.f4;
}

// ---------------------------------------------------------------------------
// Per-channel triangle matmul.  grid = (4 iTiles, 4 jTiles, 128 channels),
// 256 threads (8 waves), each wave 2 output 16x16 tiles, K staged in LDS.
//   outgoing: out[i,j] = sum_k I[i,k]*J[j,k]   (async DMA staging)
//   incoming: out[i,j] = sum_k I[k,i]*J[k,j]   (transposed manual staging)
// ---------------------------------------------------------------------------
__global__ __launch_bounds__(256) void tri_mm_kernel(
    const _Float16* __restrict__ i_t, const _Float16* __restrict__ j_t,
    float* __restrict__ mm, int outgoing) {
  __shared__ __align__(16) _Float16 sI[64][32];
  __shared__ __align__(16) _Float16 sJ[64][32];
  const int i0 = blockIdx.x * 64, j0 = blockIdx.y * 64, c = blockIdx.z;
  const int tid = threadIdx.x, wave = tid >> 5, lane = tid & 31;
  const int n = lane & 15, hi = lane >> 4;
  const int ti = (wave * 2) >> 2, tj0 = (wave * 2) & 3, tj1 = tj0 + 1;
  const size_t cb = (size_t)c * PPOS;
  v8f accA{}, accB{};
  for (int kc = 0; kc < 8; ++kc) {
    if (outgoing) {
      int r = tid & 63, seg = tid >> 6;  // 64 rows x 4 segs of 8 halfs
      async_load_b128((unsigned)(unsigned long long)(void*)&sI[r][seg * 8],
                      i_t + cb + (size_t)(i0 + r) * NNPX + kc * 32 + seg * 8);
      async_load_b128((unsigned)(unsigned long long)(void*)&sJ[r][seg * 8],
                      j_t + cb + (size_t)(j0 + r) * NNPX + kc * 32 + seg * 8);
      wait_async0();
    } else {
      int kk = tid & 31, seg = tid >> 5;  // transpose on load
      union { f4raw f4; _Float16 hh[8]; } ui, uj;
      ui.f4 = *(const f4raw*)(i_t + cb + (size_t)(kc * 32 + kk) * NNPX + i0 + seg * 8);
      uj.f4 = *(const f4raw*)(j_t + cb + (size_t)(kc * 32 + kk) * NNPX + j0 + seg * 8);
      for (int q = 0; q < 8; ++q) {
        sI[seg * 8 + q][kk] = ui.hh[q];
        sJ[seg * 8 + q][kk] = uj.hh[q];
      }
    }
    __syncthreads();
    v16h a = load_a_frag(&sI[ti * 16][0], 32, lane);
    v16h b0 = load_b_frag(&sJ[tj0 * 16][0], 32, lane);
    v16h b1 = load_b_frag(&sJ[tj1 * 16][0], 32, lane);
    accA = wmma16(a, b0, accA);
    accB = wmma16(a, b1, accB);
    __syncthreads();
  }
#pragma unroll
  for (int r = 0; r < 8; ++r) {
    int gi = i0 + ti * 16 + hi * 8 + r;
    mm[((size_t)gi * NNPX + j0 + tj0 * 16 + n) * CCH + c] = accA[r];
    mm[((size_t)gi * NNPX + j0 + tj1 * 16 + n) * CCH + c] = accB[r];
  }
}

// final: x += (LN2(mm)@Wp + bp) * sigmoid(xn@Ws + bs)
__global__ __launch_bounds__(32) void mult_final_kernel(
    const _Float16* __restrict__ mmn, const _Float16* __restrict__ Wp_t,
    const float* __restrict__ bp, const _Float16* __restrict__ xn,
    const _Float16* __restrict__ Ws_t, const float* __restrict__ bs,
    float* __restrict__ x) {
  int lane = threadIdx.x, p0 = blockIdx.x * 16, ct = blockIdx.y;
  int n = lane & 15, hi = lane >> 4;
  v8f acc{}, accs{};
  for (int kk = 0; kk < 4; ++kk) {
    v16h a1 = load_a_frag(mmn + (size_t)p0 * CCH + kk * 32, CCH, lane);
    v16h a2 = load_a_frag(xn + (size_t)p0 * CCH + kk * 32, CCH, lane);
    v16h b1 = load_b_frag(Wp_t + (size_t)(ct * 16) * CCH + kk * 32, CCH, lane);
    v16h b2 = load_b_frag(Ws_t + (size_t)(ct * 16) * CCH + kk * 32, CCH, lane);
    acc = wmma16(a1, b1, acc);
    accs = wmma16(a2, b2, accs);
  }
  int col = ct * 16 + n;
  float bpv = bp[col], bsv = bs[col];
  for (int r = 0; r < 8; ++r) {
    size_t p = p0 + hi * 8 + r;
    x[p * CCH + col] += (acc[r] + bpv) * sigm(accs[r] + bsv);
  }
}

// ---------------------------------------------------------------------------
// Host side
// ---------------------------------------------------------------------------
#define MB1 (1024ull * 1024ull)

struct WSP {
  _Float16 *xn16, *bufA, *bufB, *bufC, *g16, *o16;
  float *mm, *bias_t;
  _Float16 *wqkv_t, *wg_t, *wo_t, *w1, *w2, *w3, *w4, *w5, *w6;
};

static inline void launch_wt(const float* W, _Float16* Wt, int K, int Nn,
                             hipStream_t s) {
  int elems = K * Nn;
  wt_kernel<<<(elems + 255) / 256, 256, 0, s>>>(W, Wt, K, Nn);
}

static void run_attn(void* const* in, int b, const float* xin, float* xout,
                     const WSP& w, int starting, hipStream_t stream) {
  const float* ln_s = (const float*)in[b + 0];
  const float* ln_b = (const float*)in[b + 1];
  const float* Wqkv = (const float*)in[b + 2];
  const float* Wb   = (const float*)in[b + 3];
  const float* Wg   = (const float*)in[b + 4];
  const float* bg   = (const float*)in[b + 5];
  const float* Wo   = (const float*)in[b + 6];
  const float* bo   = (const float*)in[b + 7];
  ln_kernel<<<PPOS, 32, 0, stream>>>(xin, ln_s, ln_b, w.xn16);
  launch_wt(Wqkv, w.wqkv_t, 128, 384, stream);
  launch_wt(Wg, w.wg_t, 128, 128, stream);
  launch_wt(Wo, w.wo_t, 128, 128, stream);
  attn_proj_kernel<<<dim3(4096, 32), 32, 0, stream>>>(
      w.xn16, w.wqkv_t, w.wg_t, bg, w.bufA, w.bufB, w.bufC, w.g16);
  attn_bias_kernel<<<PPOS * 4 / 256, 256, 0, stream>>>(w.xn16, Wb, w.bias_t);
  attn_core_kernel<<<dim3(256, 4), 256, 0, stream>>>(
      w.bufA, w.bufB, w.bufC, w.bias_t, w.g16, w.o16, starting);
  attn_out_kernel<<<dim3(4096, 8), 32, 0, stream>>>(w.o16, w.wo_t, bo, xin, xout);
}

static void run_mult(void* const* in, int b, float* x, const WSP& w,
                     int outgoing, hipStream_t stream) {
  const float *ln1_s = (const float*)in[b + 0], *ln1_b = (const float*)in[b + 1];
  const float *Wi = (const float*)in[b + 2],  *bi = (const float*)in[b + 3];
  const float *Wj = (const float*)in[b + 4],  *bj = (const float*)in[b + 5];
  const float *Wig = (const float*)in[b + 6], *big = (const float*)in[b + 7];
  const float *Wjg = (const float*)in[b + 8], *bjg = (const float*)in[b + 9];
  const float *ln2_s = (const float*)in[b + 10], *ln2_b = (const float*)in[b + 11];
  const float *Wp = (const float*)in[b + 12], *bp = (const float*)in[b + 13];
  const float *Ws = (const float*)in[b + 14], *bs = (const float*)in[b + 15];
  ln_kernel<<<PPOS, 32, 0, stream>>>(x, ln1_s, ln1_b, w.xn16);
  launch_wt(Wi, w.w1, 128, 128, stream);
  launch_wt(Wig, w.w2, 128, 128, stream);
  launch_wt(Wj, w.w3, 128, 128, stream);
  launch_wt(Wjg, w.w4, 128, 128, stream);
  launch_wt(Wp, w.w5, 128, 128, stream);
  launch_wt(Ws, w.w6, 128, 128, stream);
  mult_proj_kernel<<<dim3(4096, 8), 32, 0, stream>>>(w.xn16, w.w1, bi, w.w2, big, w.bufA);
  mult_proj_kernel<<<dim3(4096, 8), 32, 0, stream>>>(w.xn16, w.w3, bj, w.w4, bjg, w.bufB);
  tri_mm_kernel<<<dim3(4, 4, 128), 256, 0, stream>>>(w.bufA, w.bufB, w.mm, outgoing);
  ln_kernel<<<PPOS, 32, 0, stream>>>(w.mm, ln2_s, ln2_b, w.bufC);
  mult_final_kernel<<<dim3(4096, 8), 32, 0, stream>>>(w.bufC, w.w5, bp, w.xn16,
                                                      w.w6, bs, x);
}

extern "C" void kernel_launch(void* const* d_in, const int* in_sizes, int n_in,
                              void* d_out, int out_size, void* d_ws, size_t ws_size,
                              hipStream_t stream) {
  (void)in_sizes; (void)n_in; (void)out_size; (void)ws_size;
  char* p = (char*)d_ws;
  WSP w;
  w.xn16   = (_Float16*)(p + 0 * MB1);    // 16 MB  LN'ed activations (f16)
  w.bufA   = (_Float16*)(p + 16 * MB1);   // 16 MB  q_t   / i_t
  w.bufB   = (_Float16*)(p + 32 * MB1);   // 16 MB  k_t   / j_t
  w.bufC   = (_Float16*)(p + 48 * MB1);   // 16 MB  v_t   / LN2(mm) f16
  w.g16    = (_Float16*)(p + 64 * MB1);   // 16 MB  attn gate (attn phase)
  w.o16    = (_Float16*)(p + 80 * MB1);   // 16 MB  attn output (attn phase)
  w.mm     = (float*)(p + 64 * MB1);      // 32 MB  triangle matmul out (mult phase)
  w.bias_t = (float*)(p + 96 * MB1);      // 1 MB   attn bias [h][p]
  char* wa = p + 97 * MB1;                // transposed f16 weight arena
  w.wqkv_t = (_Float16*)(wa + 0);
  w.wg_t   = (_Float16*)(wa + 96 * 1024);
  w.wo_t   = (_Float16*)(wa + 128 * 1024);
  w.w1     = (_Float16*)(wa + 160 * 1024);
  w.w2     = (_Float16*)(wa + 192 * 1024);
  w.w3     = (_Float16*)(wa + 224 * 1024);
  w.w4     = (_Float16*)(wa + 256 * 1024);
  w.w5     = (_Float16*)(wa + 288 * 1024);
  w.w6     = (_Float16*)(wa + 320 * 1024);

  const float* x_in = (const float*)d_in[0];
  float* x = (float*)d_out;

  // input order: x2d, attn_start{ln_s,ln_b,Wqkv,Wb,Wg,bg,Wo,bo},
  //              attn_end{..8..}, mult_out{..16..}, mult_in{..16..}
  run_attn(d_in, 1, x_in, x, w, /*starting=*/1, stream);  // x = x_in + attn_start
  run_attn(d_in, 9, x, x, w, /*starting=*/0, stream);     // x += attn_end
  run_mult(d_in, 17, x, w, /*outgoing=*/1, stream);       // x += mult_out
  run_mult(d_in, 33, x, w, /*outgoing=*/0, stream);       // x += mult_in
}
